// Decoder_83794811945681
// MI455X (gfx1250) — matrix operand order repaired and checked by reading, not compiled, not run on hardware
//
#include <hip/hip_runtime.h>

// ============================================================================
// GraphCast M2G decoder on MI455X (gfx1250).
// All GEMMs run on v_wmma_f32_16x16x32_bf16 (fp32 accumulate).
// Weights are pre-swizzled once into the WMMA B-fragment lane layout so each
// B fragment is two contiguous 16B loads per lane. Activations staged in LDS
// as bf16 with conflict-free strides; LayerNorm via __shfl_xor cross-lane
// reduction (wave32). Edge scatter uses global_atomic_add_f32 into agg.
// ============================================================================

typedef __bf16 bf16_t;
typedef bf16_t v16bf __attribute__((ext_vector_type(16)));
typedef float  v8f   __attribute__((ext_vector_type(8)));

struct U4 { unsigned int a, b, c, d; } __attribute__((aligned(16)));
struct F4 { float x, y, z, w; }        __attribute__((aligned(16)));

#define H 128
#define WAVES 4
// LDS strides in elements; chosen so row stride (bytes) % 256 == 16 (bank
// shift of 4 per row) and % 16 == 0 (aligned ds_read_b128).
#define A1S_EDGE 392   // 16 x 392 bf16  (edge concat input, 384 cols)
#define A1S_NODE 264   // 16 x 264 bf16  (node concat input, 256 cols)
#define A2S      136   // 16 x 136 bf16  (128-col bf16 activations)
#define HS       132   // 16 x 132 f32   (128-col fp32 activations)

union Frag { v16bf v; U4 u[2]; };

// A fragment (16x32 bf16, MxK) per ISA layout:
//   lanes 0-15: M=lane,   VGPR0..3 -> K=0..7,  VGPR4..7 -> K=16..23
//   lanes16-31: M=lane-16,VGPR0..3 -> K=8..15, VGPR4..7 -> K=24..31
__device__ __forceinline__ v16bf lds_a_frag(const bf16_t* Ab, int stride,
                                            int kt, int lane) {
  int m  = lane & 15;
  int kb = kt * 32 + ((lane & 16) ? 8 : 0);
  const bf16_t* p = Ab + m * stride + kb;
  Frag t;
  t.u[0] = *(const U4*)(p);        // K = kb .. kb+7
  t.u[1] = *(const U4*)(p + 16);   // K = kb+16 .. kb+23
  return t.v;
}

// B fragment: pre-swizzled in global memory, 32B contiguous per lane.
__device__ __forceinline__ v16bf glb_b_frag(const bf16_t* W, int ntiles,
                                            int kt, int nt, int lane) {
  const bf16_t* p = W + (((size_t)(kt * ntiles + nt) * 32 + lane) << 4);
  Frag t;
  t.u[0] = ((const U4*)p)[0];
  t.u[1] = ((const U4*)p)[1];
  return t.v;
}

__device__ __forceinline__ float silu_f(float x) {
  return x / (1.0f + __expf(-x));
}

#define WMMA_BF16(ACC, A, B) \
  (ACC) = __builtin_amdgcn_wmma_f32_16x16x32_bf16(false, (A), false, (B), \
                                                  (short)0, (ACC), false, false)

// ---------------------------------------------------------------------------
// Pre-swizzle a KxN fp32 weight matrix into bf16 B-fragment layout.
// B fragment lane layout (32x16 bf16, KxN):
//   lanes 0-15: N=lane,    elements i=0..7 -> K = kt*32 + i,     i=8..15 -> K = kt*32 + 8 + i
//   lanes16-31: N=lane-16, elements i=0..7 -> K = kt*32 + 8 + i, i=8..15 -> K = kt*32 + 16 + i
// dst[tile][lane][i], tile = kt*ntiles + nt. Zero-pads n >= N.
// ---------------------------------------------------------------------------
__global__ void swizzle_weight(const float* __restrict__ W, int K, int N,
                               int Npad, bf16_t* __restrict__ dst) {
  int ntiles = Npad >> 4;
  int tile = blockIdx.x;                   // kt*ntiles + nt
  int kt = tile / ntiles, nt = tile - kt * ntiles;
  int lane = threadIdx.x >> 4;             // 0..31
  int i    = threadIdx.x & 15;             // 0..15
  int k = kt * 32 + ((lane & 16) ? 8 : 0) + (i < 8 ? i : 8 + i);
  int n = nt * 16 + (lane & 15);
  float v = (k < K && n < N) ? W[(size_t)k * N + n] : 0.0f;
  dst[(((size_t)tile * 32 + lane) << 4) + i] = (bf16_t)v;
}

__global__ void zero_f32(float* __restrict__ p, long n) {
  long i = (long)blockIdx.x * blockDim.x + threadIdx.x;
  long stride = (long)gridDim.x * blockDim.x;
  for (; i < n; i += stride) p[i] = 0.0f;
}

// ---------------------------------------------------------------------------
// Edge kernel: one wave = 16 edges of one batch.
//   h  = LN(silu([eM2G | vM[s] | vG[r]] @ w_e1 + b_e1))
//   e  = eM2G + h @ w_e2 + b_e2
//   agg[b, r] += e  (atomic scatter)
// ---------------------------------------------------------------------------
__global__ __launch_bounds__(WAVES * 32)
void edge_kernel(const float* __restrict__ eM2G, const float* __restrict__ vM,
                 const float* __restrict__ vG,
                 const int* __restrict__ senders,
                 const int* __restrict__ receivers,
                 const bf16_t* __restrict__ w1, const float* __restrict__ b1,
                 const float* __restrict__ gln, const float* __restrict__ bln,
                 const bf16_t* __restrict__ w2, const float* __restrict__ b2,
                 float* __restrict__ agg,
                 int E, int Nm, int Ng, int totalTiles) {
  extern __shared__ __attribute__((aligned(16))) char smem[];
  const int wave = threadIdx.x >> 5;
  const int lane = threadIdx.x & 31;
  bf16_t* Abuf = (bf16_t*)smem + wave * (16 * A1S_EDGE);
  float*  Hbuf = (float*)(smem + WAVES * 16 * A1S_EDGE * 2) + wave * (16 * HS);

  int tile = blockIdx.x * WAVES + wave;
  const bool valid = tile < totalTiles;
  if (!valid) tile = 0;
  const int tilesPerB = E >> 4;
  const int b  = tile / tilesPerB;
  const int e0 = (tile - b * tilesPerB) << 4;
  const int nlo   = lane & 15;
  const int half8 = (lane & 16) ? 8 : 0;

  // ---- stage A1 = [eM2G | vM[s] | vG[r]]  (16 x 384 bf16)
  {
    int m = lane & 15;
    int e = e0 + m;
    int s = senders[e];
    int r = receivers[e];
    const float* s0 = eM2G + (size_t)e * H;
    const float* s1 = vM + ((size_t)b * Nm + s) * H;
    const float* s2 = vG + ((size_t)b * Ng + r) * H;
    bf16_t* drow = Abuf + m * A1S_EDGE;
    int c0 = (lane & 16) ? 192 : 0;
    for (int c = c0; c < c0 + 192; c += 4) {
      const float* sp = (c < 128) ? (s0 + c)
                      : ((c < 256) ? (s1 + (c - 128)) : (s2 + (c - 256)));
      F4 f = *(const F4*)sp;
      drow[c + 0] = (bf16_t)f.x; drow[c + 1] = (bf16_t)f.y;
      drow[c + 2] = (bf16_t)f.z; drow[c + 3] = (bf16_t)f.w;
    }
  }
  __syncthreads();

  const v8f vzero = {0.f, 0.f, 0.f, 0.f, 0.f, 0.f, 0.f, 0.f};

  // ---- GEMM1: (16x384) @ (384x128), 12 k-steps x 8 n-tiles
  v8f acc[8];
#pragma unroll
  for (int nt = 0; nt < 8; ++nt) acc[nt] = vzero;
  for (int kt = 0; kt < 12; ++kt) {
    v16bf a = lds_a_frag(Abuf, A1S_EDGE, kt, lane);
#pragma unroll
    for (int nt = 0; nt < 8; ++nt) {
      v16bf bb = glb_b_frag(w1, 8, kt, nt, lane);
      WMMA_BF16(acc[nt], a, bb);
    }
  }

  // ---- bias + SiLU -> Hbuf (f32)
#pragma unroll
  for (int nt = 0; nt < 8; ++nt) {
    float bias = b1[nt * 16 + nlo];
#pragma unroll
    for (int j = 0; j < 8; ++j)
      Hbuf[(j + half8) * HS + nt * 16 + nlo] = silu_f(acc[nt][j] + bias);
  }
  __syncthreads();

  // ---- LayerNorm over 128 features (2 lanes per row, combine via shfl_xor 16)
  {
    int m = lane & 15;
    int cseg = (lane & 16) ? 64 : 0;
    const float* row = Hbuf + m * HS;
    float s = 0.f, sq = 0.f;
    for (int c = cseg; c < cseg + 64; ++c) { float v = row[c]; s += v; sq += v * v; }
    s  += __shfl_xor(s, 16, 32);
    sq += __shfl_xor(sq, 16, 32);
    float mean = s * (1.0f / 128.0f);
    float var  = sq * (1.0f / 128.0f) - mean * mean;
    float rstd = rsqrtf(var + 1e-5f);
    bf16_t* arow = Abuf + m * A2S;   // reuse Abuf region as A2 (16x128 bf16)
    for (int c = cseg; c < cseg + 64; ++c)
      arow[c] = (bf16_t)((row[c] - mean) * rstd * gln[c] + bln[c]);
  }
  __syncthreads();

  // ---- GEMM2: (16x128) @ (128x128)
  v8f acc2[8];
#pragma unroll
  for (int nt = 0; nt < 8; ++nt) acc2[nt] = vzero;
  for (int kt = 0; kt < 4; ++kt) {
    v16bf a = lds_a_frag(Abuf, A2S, kt, lane);
#pragma unroll
    for (int nt = 0; nt < 8; ++nt) {
      v16bf bb = glb_b_frag(w2, 8, kt, nt, lane);
      WMMA_BF16(acc2[nt], a, bb);
    }
  }

  // ---- residual (+eM2G) and atomic scatter into agg[b, receivers[e]]
  if (valid) {
#pragma unroll
    for (int j = 0; j < 8; ++j) {
      int e = e0 + half8 + j;
      int r = receivers[e];
      const float* esrc = eM2G + (size_t)e * H;
      float* dst = agg + ((size_t)b * Ng + r) * H;
#pragma unroll
      for (int nt = 0; nt < 8; ++nt) {
        int n = nt * 16 + nlo;
        atomicAdd(dst + n, acc2[nt][j] + b2[n] + esrc[n]);
      }
    }
  }
}

// ---------------------------------------------------------------------------
// Fused node + head kernel: one wave = 16 grid nodes of one batch.
//   h   = LN(silu([vG | agg] @ w_n1 + b_n1));  vg = vG + h @ w_n2 + b_n2
//   out = silu(vg @ w_h1 + b_h1) @ w_h2 + b_h2
// ---------------------------------------------------------------------------
__global__ __launch_bounds__(WAVES * 32)
void node_head_kernel(const float* __restrict__ vG, const float* __restrict__ agg,
                      const bf16_t* __restrict__ w1, const float* __restrict__ b1,
                      const float* __restrict__ gln, const float* __restrict__ bln,
                      const bf16_t* __restrict__ w2, const float* __restrict__ b2,
                      const bf16_t* __restrict__ wh1, const float* __restrict__ bh1,
                      const bf16_t* __restrict__ wh2, const float* __restrict__ bh2,
                      float* __restrict__ out, int Ng, int totalTiles) {
  extern __shared__ __attribute__((aligned(16))) char smem[];
  const int wave = threadIdx.x >> 5;
  const int lane = threadIdx.x & 31;
  bf16_t* Abuf = (bf16_t*)smem + wave * (16 * A1S_NODE);
  float*  Hbuf = (float*)(smem + WAVES * 16 * A1S_NODE * 2) + wave * (16 * HS);

  int tile = blockIdx.x * WAVES + wave;
  const bool valid = tile < totalTiles;
  if (!valid) tile = 0;
  const int tilesPerB = Ng >> 4;
  const int b  = tile / tilesPerB;
  const int g0 = (tile - b * tilesPerB) << 4;
  const int nlo   = lane & 15;
  const int half8 = (lane & 16) ? 8 : 0;

  // ---- stage A1 = [vG | agg]  (16 x 256 bf16)
  {
    int m = lane & 15;
    size_t rowidx = ((size_t)b * Ng + g0 + m) * H;
    const float* src = (lane & 16) ? (agg + rowidx) : (vG + rowidx);
    int c0 = (lane & 16) ? 128 : 0;
    bf16_t* drow = Abuf + m * A1S_NODE;
    for (int c = 0; c < 128; c += 4) {
      F4 f = *(const F4*)(src + c);
      drow[c0 + c + 0] = (bf16_t)f.x; drow[c0 + c + 1] = (bf16_t)f.y;
      drow[c0 + c + 2] = (bf16_t)f.z; drow[c0 + c + 3] = (bf16_t)f.w;
    }
  }
  __syncthreads();

  const v8f vzero = {0.f, 0.f, 0.f, 0.f, 0.f, 0.f, 0.f, 0.f};

  // ---- GEMM1: (16x256) @ (256x128)
  v8f acc[8];
#pragma unroll
  for (int nt = 0; nt < 8; ++nt) acc[nt] = vzero;
  for (int kt = 0; kt < 8; ++kt) {
    v16bf a = lds_a_frag(Abuf, A1S_NODE, kt, lane);
#pragma unroll
    for (int nt = 0; nt < 8; ++nt) {
      v16bf bb = glb_b_frag(w1, 8, kt, nt, lane);
      WMMA_BF16(acc[nt], a, bb);
    }
  }
#pragma unroll
  for (int nt = 0; nt < 8; ++nt) {
    float bias = b1[nt * 16 + nlo];
#pragma unroll
    for (int j = 0; j < 8; ++j)
      Hbuf[(j + half8) * HS + nt * 16 + nlo] = silu_f(acc[nt][j] + bias);
  }
  __syncthreads();

  // ---- LayerNorm -> A2 (bf16, in Abuf)
  {
    int m = lane & 15;
    int cseg = (lane & 16) ? 64 : 0;
    const float* row = Hbuf + m * HS;
    float s = 0.f, sq = 0.f;
    for (int c = cseg; c < cseg + 64; ++c) { float v = row[c]; s += v; sq += v * v; }
    s  += __shfl_xor(s, 16, 32);
    sq += __shfl_xor(sq, 16, 32);
    float mean = s * (1.0f / 128.0f);
    float var  = sq * (1.0f / 128.0f) - mean * mean;
    float rstd = rsqrtf(var + 1e-5f);
    bf16_t* arow = Abuf + m * A2S;
    for (int c = cseg; c < cseg + 64; ++c)
      arow[c] = (bf16_t)((row[c] - mean) * rstd * gln[c] + bln[c]);
  }
  __syncthreads();

  // ---- GEMM2: (16x128) @ (128x128)
  v8f acc2[8];
#pragma unroll
  for (int nt = 0; nt < 8; ++nt) acc2[nt] = vzero;
  for (int kt = 0; kt < 4; ++kt) {
    v16bf a = lds_a_frag(Abuf, A2S, kt, lane);
#pragma unroll
    for (int nt = 0; nt < 8; ++nt) {
      v16bf bb = glb_b_frag(w2, 8, kt, nt, lane);
      WMMA_BF16(acc2[nt], a, bb);
    }
  }
  __syncthreads();

  // ---- vg = vG + acc2 + b_n2 -> A3 (bf16, reuse Abuf)
#pragma unroll
  for (int j = 0; j < 8; ++j) {
    int m = half8 + j;
    const float* vrow = vG + ((size_t)b * Ng + g0 + m) * H;
    bf16_t* arow = Abuf + m * A2S;
#pragma unroll
    for (int nt = 0; nt < 8; ++nt) {
      int n = nt * 16 + nlo;
      arow[n] = (bf16_t)(acc2[nt][j] + b2[n] + vrow[n]);
    }
  }
  __syncthreads();

  // ---- GEMM3: (16x128) @ (128x128) head layer 1
  v8f acc3[8];
#pragma unroll
  for (int nt = 0; nt < 8; ++nt) acc3[nt] = vzero;
  for (int kt = 0; kt < 4; ++kt) {
    v16bf a = lds_a_frag(Abuf, A2S, kt, lane);
#pragma unroll
    for (int nt = 0; nt < 8; ++nt) {
      v16bf bb = glb_b_frag(wh1, 8, kt, nt, lane);
      WMMA_BF16(acc3[nt], a, bb);
    }
  }
  // silu -> A4 (bf16, reuse Hbuf region)
  bf16_t* H16 = (bf16_t*)Hbuf;
#pragma unroll
  for (int nt = 0; nt < 8; ++nt) {
    float bias = bh1[nt * 16 + nlo];
#pragma unroll
    for (int j = 0; j < 8; ++j)
      H16[(j + half8) * A2S + nt * 16 + nlo] = (bf16_t)silu_f(acc3[nt][j] + bias);
  }
  __syncthreads();

  // ---- GEMM4: (16x128) @ (128x16 zero-padded, only 8 cols valid)
  v8f accO = vzero;
  for (int kt = 0; kt < 4; ++kt) {
    v16bf a  = lds_a_frag(H16, A2S, kt, lane);
    v16bf bb = glb_b_frag(wh2, 1, kt, 0, lane);
    WMMA_BF16(accO, a, bb);
  }

  // ---- write out[b, g0+m, 0..7]
  if (valid && nlo < 8) {
    float bias = bh2[nlo];
#pragma unroll
    for (int j = 0; j < 8; ++j)
      out[((size_t)b * Ng + g0 + half8 + j) * 8 + nlo] = accO[j] + bias;
  }
}

// ---------------------------------------------------------------------------
extern "C" void kernel_launch(void* const* d_in, const int* in_sizes, int n_in,
                              void* d_out, int out_size, void* d_ws, size_t ws_size,
                              hipStream_t stream) {
  const float* vM   = (const float*)d_in[0];
  const float* vG   = (const float*)d_in[1];
  const float* eM2G = (const float*)d_in[2];
  const int* senders   = (const int*)d_in[3];
  const int* receivers = (const int*)d_in[4];
  const float* w_e1 = (const float*)d_in[5];
  const float* b_e1 = (const float*)d_in[6];
  const float* g_e  = (const float*)d_in[7];
  const float* be_e = (const float*)d_in[8];
  const float* w_e2 = (const float*)d_in[9];
  const float* b_e2 = (const float*)d_in[10];
  const float* w_n1 = (const float*)d_in[11];
  const float* b_n1 = (const float*)d_in[12];
  const float* g_n  = (const float*)d_in[13];
  const float* be_n = (const float*)d_in[14];
  const float* w_n2 = (const float*)d_in[15];
  const float* b_n2 = (const float*)d_in[16];
  const float* w_h1 = (const float*)d_in[17];
  const float* b_h1 = (const float*)d_in[18];
  const float* w_h2 = (const float*)d_in[19];
  const float* b_h2 = (const float*)d_in[20];

  const int B  = 2;
  const int Nm = in_sizes[0] / (B * H);
  const int Ng = in_sizes[1] / (B * H);
  const int E  = in_sizes[3];

  // workspace layout: [agg (B*Ng*H f32)] [swizzled bf16 weights]
  float* agg = (float*)d_ws;
  size_t aggBytes = (size_t)B * Ng * H * sizeof(float);
  char* wsp = (char*)d_ws + ((aggBytes + 255) & ~(size_t)255);
  bf16_t* w_e1_sw = (bf16_t*)wsp;              // 12x8 tiles
  bf16_t* w_e2_sw = w_e1_sw + 12 * 8 * 512;    //  4x8
  bf16_t* w_n1_sw = w_e2_sw + 4 * 8 * 512;     //  8x8
  bf16_t* w_n2_sw = w_n1_sw + 8 * 8 * 512;     //  4x8
  bf16_t* w_h1_sw = w_n2_sw + 4 * 8 * 512;     //  4x8
  bf16_t* w_h2_sw = w_h1_sw + 4 * 8 * 512;     //  4x1 (N padded 8->16)

  swizzle_weight<<<dim3(12 * 8), dim3(512), 0, stream>>>(w_e1, 384, 128, 128, w_e1_sw);
  swizzle_weight<<<dim3(4 * 8),  dim3(512), 0, stream>>>(w_e2, 128, 128, 128, w_e2_sw);
  swizzle_weight<<<dim3(8 * 8),  dim3(512), 0, stream>>>(w_n1, 256, 128, 128, w_n1_sw);
  swizzle_weight<<<dim3(4 * 8),  dim3(512), 0, stream>>>(w_n2, 128, 128, 128, w_n2_sw);
  swizzle_weight<<<dim3(4 * 8),  dim3(512), 0, stream>>>(w_h1, 128, 128, 128, w_h1_sw);
  swizzle_weight<<<dim3(4),      dim3(512), 0, stream>>>(w_h2, 128, 8, 16, w_h2_sw);

  zero_f32<<<dim3(2048), dim3(256), 0, stream>>>(agg, (long)B * Ng * H);

  const int eTiles  = B * (E >> 4);
  const int eBlocks = (eTiles + WAVES - 1) / WAVES;
  const size_t eSmem = (size_t)WAVES * (16 * A1S_EDGE * 2 + 16 * HS * 4);
  edge_kernel<<<dim3(eBlocks), dim3(WAVES * 32), eSmem, stream>>>(
      eM2G, vM, vG, senders, receivers,
      w_e1_sw, b_e1, g_e, be_e, w_e2_sw, b_e2, agg, E, Nm, Ng, eTiles);

  const int nTiles  = B * (Ng >> 4);
  const int nBlocks = (nTiles + WAVES - 1) / WAVES;
  const size_t nSmem = (size_t)WAVES * (16 * A1S_NODE * 2 + 16 * HS * 4);
  node_head_kernel<<<dim3(nBlocks), dim3(WAVES * 32), nSmem, stream>>>(
      vG, agg, w_n1_sw, b_n1, g_n, be_n, w_n2_sw, b_n2,
      w_h1_sw, b_h1, w_h2_sw, b_h2, (float*)d_out, Ng, nTiles);
}
